// DynamicMLP_89395449299721
// MI455X (gfx1250) — compile-verified, matrix operand order checked by reading, not run on hardware
//
#include <hip/hip_runtime.h>
#include <hip/hip_bf16.h>
#include <math.h>

typedef __attribute__((ext_vector_type(16))) __bf16 v16bf;
typedef __attribute__((ext_vector_type(8)))  __bf16 v8bf;
typedef __attribute__((ext_vector_type(4)))  __bf16 v4bf;
typedef __attribute__((ext_vector_type(2)))  __bf16 v2bf;
typedef __attribute__((ext_vector_type(8)))  float  v8f;
typedef __attribute__((ext_vector_type(4)))  float  v4f;

#define BB   2
#define TT   2048
#define HH   1024
#define II   2048
#define EE   8
#define NTOK (BB*TT)

#define MT   128
#define NT   128
#define KT   64
#define LDK  72   // bf16 row stride: 144 bytes = 36 banks -> conflict-free frag reads

static __device__ __forceinline__ v8f zero8() {
  v8f z = {0.f,0.f,0.f,0.f,0.f,0.f,0.f,0.f};
  return z;
}

static __device__ __forceinline__ v16bf make_frag(const __bf16* base) {
  v8bf lo = *(const v8bf*)(base);
  v8bf hi = *(const v8bf*)(base + 16);
  v16bf f;
#pragma unroll
  for (int e = 0; e < 8; ++e) { f[e] = lo[e]; f[8+e] = hi[e]; }
  return f;
}

// ---------------------------------------------------------------------------
// Router: one wave32 per token. RMSNorm -> proto attn -> softmax -> value mix
// -> softmax -> top-2 renormalize. Writes rw to d_out's second region.
// ---------------------------------------------------------------------------
__global__ __launch_bounds__(256) void router_kernel(
    const float* __restrict__ x, const float* __restrict__ qw,
    const float* __restrict__ kproto, const float* __restrict__ vproto,
    float* __restrict__ rw_out)
{
  const int tok  = (blockIdx.x * blockDim.x + threadIdx.x) >> 5;
  const int lane = threadIdx.x & 31;
  if (tok >= NTOK) return;

  const float* xt = x + (size_t)tok * HH;
  float ss = 0.f;
  float dot[EE];
#pragma unroll
  for (int p = 0; p < EE; ++p) dot[p] = 0.f;

  for (int h = lane; h < HH; h += 32) {
    float xv = xt[h];
    ss += xv * xv;
    float q = xv * qw[h];
#pragma unroll
    for (int p = 0; p < EE; ++p) dot[p] += q * kproto[p*HH + h];
  }
#pragma unroll
  for (int m = 16; m >= 1; m >>= 1) {
    ss += __shfl_xor(ss, m, 32);
#pragma unroll
    for (int p = 0; p < EE; ++p) dot[p] += __shfl_xor(dot[p], m, 32);
  }

  const float rinv  = rsqrtf(ss * (1.f/(float)HH) + 1e-6f);
  const float scale = rinv * rsqrtf((float)HH);   // rinv / sqrt(H)

  float attn[EE];
  float mx = -1e30f;
#pragma unroll
  for (int p = 0; p < EE; ++p) { attn[p] = dot[p] * scale; mx = fmaxf(mx, attn[p]); }
  float s = 0.f;
#pragma unroll
  for (int p = 0; p < EE; ++p) { attn[p] = __expf(attn[p] - mx); s += attn[p]; }
  float inv = 1.f / s;
#pragma unroll
  for (int p = 0; p < EE; ++p) attn[p] *= inv;

  float lg[EE];
#pragma unroll
  for (int q = 0; q < EE; ++q) {
    float a = 0.f;
#pragma unroll
    for (int p = 0; p < EE; ++p) a += attn[p] * vproto[p*EE + q];
    lg[q] = a; // ROUTING_GAIN == 1.0
  }

  float rwv[EE];
  mx = -1e30f;
#pragma unroll
  for (int p = 0; p < EE; ++p) mx = fmaxf(mx, lg[p]);
  s = 0.f;
#pragma unroll
  for (int p = 0; p < EE; ++p) { rwv[p] = __expf(lg[p] - mx); s += rwv[p]; }
  inv = 1.f / s;
#pragma unroll
  for (int p = 0; p < EE; ++p) rwv[p] *= inv;

  // top-2 (first occurrence on ties, like lax.top_k)
  int i1 = 0; float m1 = rwv[0];
#pragma unroll
  for (int p = 1; p < EE; ++p) if (rwv[p] > m1) { m1 = rwv[p]; i1 = p; }
  int i2 = -1; float m2 = -1e30f;
#pragma unroll
  for (int p = 0; p < EE; ++p) if (p != i1 && rwv[p] > m2) { m2 = rwv[p]; i2 = p; }
  const float denom = 1.f / (m1 + m2 + 1e-9f);
#pragma unroll
  for (int p = 0; p < EE; ++p)
    rwv[p] = (p == i1 || p == i2) ? rwv[p] * denom : 0.f;

  if (lane < EE) rw_out[(size_t)tok*EE + lane] = rwv[lane];
}

// ---------------------------------------------------------------------------
// Shared tile loaders
// ---------------------------------------------------------------------------
// A tile: 128 rows x 64 k, scaled by rw[t,p], f32 -> bf16, [m][k] stride LDK.
static __device__ __forceinline__ void load_a_tile(
    __bf16* As, const float* __restrict__ A, const float* rwS,
    int tid, int m0, int k0, int ldg, int p)
{
#pragma unroll
  for (int it = 0; it < 8; ++it) {
    int q   = it*256 + tid;
    int row = q >> 4;
    int c4  = q & 15;
    v4f v = *(const v4f*)(A + (size_t)(m0+row)*ldg + k0 + c4*4);
    float sc = rwS[row*EE + p];
    v4bf pk;
#pragma unroll
    for (int j = 0; j < 4; ++j) pk[j] = (__bf16)(v[j] * sc);
    *(v4bf*)(&As[row*LDK + c4*4]) = pk;
  }
}

// B tile: 64 k-rows x 128 n of W[k][n] -> transposed bf16 LDS [n][k],
// packing (k, k+1) pairs into b32 stores.
static __device__ __forceinline__ void load_b_tile(
    __bf16* Bs, const float* __restrict__ W, int tid, int k0, int n0, int ldg)
{
#pragma unroll
  for (int it = 0; it < 4; ++it) {
    int q  = it*256 + tid;
    int kp = q >> 5;            // k-pair 0..31
    int c4 = q & 31;            // n/4   0..31
    const float* src = W + (size_t)(k0 + 2*kp)*ldg + n0 + c4*4;
    v4f r0 = *(const v4f*)(src);
    v4f r1 = *(const v4f*)(src + ldg);
#pragma unroll
    for (int j = 0; j < 4; ++j) {
      v2bf pk;
      pk[0] = (__bf16)r0[j];    // k = 2*kp   (low halfword)
      pk[1] = (__bf16)r1[j];    // k = 2*kp+1 (high halfword)
      *(v2bf*)(&Bs[(c4*4 + j)*LDK + 2*kp]) = pk;
    }
  }
}

// ---------------------------------------------------------------------------
// Fused gate/up GEMM over 8 experts with routing weights folded into A rows.
// acc_g/acc_u stay in VGPRs across expert & K loops; SiLU·mul epilogue.
// C tile 128x128 per WG; 8 waves, each 32x64 via v_wmma_f32_16x16x32_bf16.
// ---------------------------------------------------------------------------
__global__ __launch_bounds__(256) void gateup_kernel(
    const float* __restrict__ x,  const float* __restrict__ Wg,
    const float* __restrict__ Wu, const float* __restrict__ rw,
    float* __restrict__ inter)
{
  __shared__ __bf16 As[MT*LDK];
  __shared__ __bf16 Bgs[NT*LDK];
  __shared__ __bf16 Bus[NT*LDK];
  __shared__ float  rwS[MT*EE];

  const int tid = threadIdx.x;
  const int m0  = blockIdx.y * MT;
  const int n0  = blockIdx.x * NT;

  for (int i = tid; i < MT*EE; i += 256) rwS[i] = rw[(size_t)m0*EE + i];
  __syncthreads();

  const int lane  = tid & 31;
  const int wv    = tid >> 5;
  const int waveM = wv & 3;              // 4 bands of 32 rows
  const int waveN = wv >> 2;             // 2 bands of 64 cols
  const int lrow  = lane & 15;
  const int khalf = (lane >> 4) << 3;    // 0 or 8

  v8f accG[2][4], accU[2][4];
#pragma unroll
  for (int i = 0; i < 2; ++i)
#pragma unroll
    for (int j = 0; j < 4; ++j) { accG[i][j] = zero8(); accU[i][j] = zero8(); }

  for (int p = 0; p < EE; ++p) {
    const float* wg = Wg + (size_t)p * HH * II;
    const float* wu = Wu + (size_t)p * HH * II;
    for (int k0 = 0; k0 < HH; k0 += KT) {
      load_a_tile(As, x, rwS, tid, m0, k0, HH, p);
      load_b_tile(Bgs, wg, tid, k0, n0, II);
      load_b_tile(Bus, wu, tid, k0, n0, II);
      __syncthreads();

#pragma unroll
      for (int ks = 0; ks < KT; ks += 32) {
        v16bf aF[2];
#pragma unroll
        for (int mA = 0; mA < 2; ++mA)
          aF[mA] = make_frag(&As[(waveM*32 + mA*16 + lrow)*LDK + ks + khalf]);

#pragma unroll
        for (int nB = 0; nB < 4; ++nB) {
          const int col = waveN*64 + nB*16 + lrow;
          v16bf bG = make_frag(&Bgs[col*LDK + ks + khalf]);
#pragma unroll
          for (int mA = 0; mA < 2; ++mA)
            accG[mA][nB] = __builtin_amdgcn_wmma_f32_16x16x32_bf16(
                false, aF[mA], false, bG, (short)0, accG[mA][nB], false, false);
          v16bf bU = make_frag(&Bus[col*LDK + ks + khalf]);
#pragma unroll
          for (int mA = 0; mA < 2; ++mA)
            accU[mA][nB] = __builtin_amdgcn_wmma_f32_16x16x32_bf16(
                false, aF[mA], false, bU, (short)0, accU[mA][nB], false, false);
        }
      }
      __syncthreads();
    }
  }

  // ---- epilogue: inter = silu(g) * u ----
  const int rofs = (lane >> 4) << 3;
#pragma unroll
  for (int mA = 0; mA < 2; ++mA)
#pragma unroll
    for (int nB = 0; nB < 4; ++nB) {
      const int c = n0 + waveN*64 + nB*16 + lrow;
#pragma unroll
      for (int j = 0; j < 8; ++j) {
        const int r = m0 + waveM*32 + mA*16 + rofs + j;
        float g = accG[mA][nB][j];
        float u = accU[mA][nB][j];
        float sig = 1.f / (1.f + __expf(-g));
        inter[(size_t)r*II + c] = g * sig * u;
      }
    }
}

// ---------------------------------------------------------------------------
// Down GEMM over 8 experts: out = sum_p (rw[:,p] ⊙ inter) @ Wd_p
// ---------------------------------------------------------------------------
__global__ __launch_bounds__(256) void down_kernel(
    const float* __restrict__ inter, const float* __restrict__ Wd,
    const float* __restrict__ rw, float* __restrict__ out)
{
  __shared__ __bf16 As[MT*LDK];
  __shared__ __bf16 Bs[NT*LDK];
  __shared__ float  rwS[MT*EE];

  const int tid = threadIdx.x;
  const int m0  = blockIdx.y * MT;
  const int n0  = blockIdx.x * NT;

  for (int i = tid; i < MT*EE; i += 256) rwS[i] = rw[(size_t)m0*EE + i];
  __syncthreads();

  const int lane  = tid & 31;
  const int wv    = tid >> 5;
  const int waveM = wv & 3;
  const int waveN = wv >> 2;
  const int lrow  = lane & 15;
  const int khalf = (lane >> 4) << 3;

  v8f acc[2][4];
#pragma unroll
  for (int i = 0; i < 2; ++i)
#pragma unroll
    for (int j = 0; j < 4; ++j) acc[i][j] = zero8();

  for (int p = 0; p < EE; ++p) {
    const float* wd = Wd + (size_t)p * II * HH;
    for (int k0 = 0; k0 < II; k0 += KT) {
      load_a_tile(As, inter, rwS, tid, m0, k0, II, p);
      load_b_tile(Bs, wd, tid, k0, n0, HH);
      __syncthreads();

#pragma unroll
      for (int ks = 0; ks < KT; ks += 32) {
        v16bf aF[2];
#pragma unroll
        for (int mA = 0; mA < 2; ++mA)
          aF[mA] = make_frag(&As[(waveM*32 + mA*16 + lrow)*LDK + ks + khalf]);

#pragma unroll
        for (int nB = 0; nB < 4; ++nB) {
          v16bf bF = make_frag(&Bs[(waveN*64 + nB*16 + lrow)*LDK + ks + khalf]);
#pragma unroll
          for (int mA = 0; mA < 2; ++mA)
            acc[mA][nB] = __builtin_amdgcn_wmma_f32_16x16x32_bf16(
                false, aF[mA], false, bF, (short)0, acc[mA][nB], false, false);
        }
      }
      __syncthreads();
    }
  }

  const int rofs = (lane >> 4) << 3;
#pragma unroll
  for (int mA = 0; mA < 2; ++mA)
#pragma unroll
    for (int nB = 0; nB < 4; ++nB) {
      const int c = n0 + waveN*64 + nB*16 + lrow;
#pragma unroll
      for (int j = 0; j < 8; ++j) {
        const int r = m0 + waveM*32 + mA*16 + rofs + j;
        out[(size_t)r*HH + c] = acc[mA][nB][j];
      }
    }
}

// ---------------------------------------------------------------------------
extern "C" void kernel_launch(void* const* d_in, const int* in_sizes, int n_in,
                              void* d_out, int out_size, void* d_ws, size_t ws_size,
                              hipStream_t stream) {
  const float* x  = (const float*)d_in[0];
  const float* qw = (const float*)d_in[1];
  const float* kp = (const float*)d_in[2];
  const float* vp = (const float*)d_in[3];
  const float* wg = (const float*)d_in[4];
  const float* wu = (const float*)d_in[5];
  const float* wd = (const float*)d_in[6];

  float* out   = (float*)d_out;
  float* rwp   = out + (size_t)NTOK * HH;   // second output region: rw [4096,8]
  float* inter = (float*)d_ws;              // NTOK*II f32 = 32 MB scratch

  router_kernel<<<NTOK/8, 256, 0, stream>>>(x, qw, kp, vp, rwp);
  gateup_kernel<<<dim3(II/NT, NTOK/MT), 256, 0, stream>>>(x, wg, wu, rwp, inter);
  down_kernel  <<<dim3(HH/NT, NTOK/MT), 256, 0, stream>>>(inter, wd, rwp, out);
}
